// Actor_7619271983217
// MI455X (gfx1250) — compile-verified, matrix operand order checked by reading, not run on hardware
//
#include <hip/hip_runtime.h>
#include <hip/hip_bf16.h>

typedef __attribute__((ext_vector_type(16))) _Float16 v16h;
typedef __attribute__((ext_vector_type(8)))  _Float16 v8h;
typedef __attribute__((ext_vector_type(8)))  float    v8f;

// ---------------------------------------------------------------------------
// helpers
// ---------------------------------------------------------------------------
__device__ __forceinline__ v8f wmma16(v16h a, v16h b, v8f c) {
  // D = A(16x32 f16) * B(32x16 f16) + C(16x16 f32)
  return __builtin_amdgcn_wmma_f32_16x16x32_f16(false, a, false, b, (short)0, c,
                                                false, false);
}

// Load one 16x32 f16 WMMA A/B fragment from row-major memory (global or LDS).
// ISA layout: lanes 0-15 hold row (lane&15), K halves {0..7,16..23};
//             lanes 16-31 hold same rows,   K halves {8..15,24..31}.
__device__ __forceinline__ v16h load_frag(const _Float16* base, int ld, int row,
                                          int kofs, int lane) {
  int r  = row + (lane & 15);
  int kb = kofs + ((lane >> 4) << 3);
  const _Float16* p = base + (size_t)r * ld + kb;
  v8h lo = *(const v8h*)(p);
  v8h hi = *(const v8h*)(p + 16);
  v16h f;
#pragma unroll
  for (int i = 0; i < 8; ++i) { f[i] = lo[i]; f[i + 8] = hi[i]; }
  return f;
}

__device__ __forceinline__ float sigm(float x) { return 1.f / (1.f + __expf(-x)); }

// ---------------------------------------------------------------------------
// f32 -> f16 conversion
// ---------------------------------------------------------------------------
__global__ void cvt_f32_f16(const float* __restrict__ in,
                            _Float16* __restrict__ out, int n) {
  int i = blockIdx.x * blockDim.x + threadIdx.x;
  if (i < n) out[i] = (_Float16)in[i];
}

// ---------------------------------------------------------------------------
// s_emb = relu(states @ se_W^T + se_b)   [8192,256] -> f16
// 8 waves/block, 1 tile/wave, tiles = 512x16
// ---------------------------------------------------------------------------
__global__ __launch_bounds__(256) void semb_gemm(const _Float16* __restrict__ A,
                                                 const _Float16* __restrict__ W,
                                                 const float* __restrict__ bias,
                                                 _Float16* __restrict__ out) {
  int lane = threadIdx.x & 31, wv = threadIdx.x >> 5;
  int tile = blockIdx.x * 8 + wv;
  int mt = tile >> 4, nt = tile & 15;
  int n = nt * 16 + (lane & 15);
  v8f acc = {};
#pragma unroll
  for (int kc = 0; kc < 8; ++kc) {
    v16h a = load_frag(A, 256, mt * 16, kc * 32, lane);
    v16h b = load_frag(W, 256, nt * 16, kc * 32, lane);
    acc = wmma16(a, b, acc);
  }
  float bv = bias[n];
  int rb = mt * 16 + ((lane >> 4) << 3);
#pragma unroll
  for (int r = 0; r < 8; ++r) {
    float v = acc[r] + bv;
    out[(size_t)(rb + r) * 256 + n] = (_Float16)(v > 0.f ? v : 0.f);
  }
}

// ---------------------------------------------------------------------------
// Gbase0 = s_emb @ Wih0[:, :256]^T + b_ih0 + b_hh0   [8192,1024] f32
// tiles = 512x64
// ---------------------------------------------------------------------------
__global__ __launch_bounds__(256) void gbase_gemm(const _Float16* __restrict__ A,
                                                  const _Float16* __restrict__ W,
                                                  const float* __restrict__ b0,
                                                  const float* __restrict__ b1,
                                                  float* __restrict__ out) {
  int lane = threadIdx.x & 31, wv = threadIdx.x >> 5;
  int tile = blockIdx.x * 8 + wv;
  int mt = tile >> 6, nt = tile & 63;
  int n = nt * 16 + (lane & 15);
  v8f acc = {};
#pragma unroll
  for (int kc = 0; kc < 8; ++kc) {
    v16h a = load_frag(A, 256, mt * 16, kc * 32, lane);
    v16h b = load_frag(W, 288, nt * 16, kc * 32, lane);  // Wih0 is [1024,288]
    acc = wmma16(a, b, acc);
  }
  float bv = b0[n] + b1[n];
  int rb = mt * 16 + ((lane >> 4) << 3);
#pragma unroll
  for (int r = 0; r < 8; ++r)
    out[(size_t)(rb + r) * 1024 + n] = acc[r] + bv;
}

// ---------------------------------------------------------------------------
// Persistent recurrent kernel: 256 blocks x 256 threads (8 waves).
// Each block owns 32 batch rows for all 32 decode steps.
// Wave w owns h-columns [32w, 32w+32) -> gate N-tiles {G*16 + 2w, G*16 + 2w+1}.
// c0/c1 stay in accumulator-layout VGPRs; h0/h1/prev live in LDS as f16.
// ---------------------------------------------------------------------------
__global__ __launch_bounds__(256) void actor_recurrent(
    const _Float16* __restrict__ Wih0h,   // [1024,288] f16 (cols 256..287 = emb part)
    const _Float16* __restrict__ Whh0h,   // [1024,256] f16
    const _Float16* __restrict__ Wih1h,   // [1024,256] f16
    const _Float16* __restrict__ Whh1h,   // [1024,256] f16
    const float*    __restrict__ Gbase0,  // [8192,1024] f32 (incl. layer0 biases)
    const float*    __restrict__ b_ih1,
    const float*    __restrict__ b_hh1,
    const _Float16* __restrict__ aembh,   // [3,32] f16
    const int*      __restrict__ actions, // [8192,32]
    const float*    __restrict__ headsW,  // [32,256,2]
    const float*    __restrict__ headsB,  // [32,2]
    float* __restrict__ tot_lp, float* __restrict__ tot_ent,
    float* __restrict__ prefix, float* __restrict__ perent) {
  __shared__ __align__(16) _Float16 sH0[32 * 256];
  __shared__ __align__(16) _Float16 sH1[32 * 256];
  __shared__ __align__(16) _Float16 sPrev[32 * 32];
  __shared__ int sAct[32];

  const int t = threadIdx.x;
  const int lane = t & 31;
  const int wv = t >> 5;
  const int rowBase = blockIdx.x * 32;
  const int colb = wv * 32;
  const int ln15 = lane & 15;
  const int rsub = (lane >> 4) << 3;

  for (int i = t; i < 32 * 256; i += 256) {
    sH0[i] = (_Float16)0.f;
    sH1[i] = (_Float16)0.f;
  }
  {  // initial prev = aemb[START_TOK=2]
    int r = t >> 3, c0 = (t & 7) * 4;
#pragma unroll
    for (int q = 0; q < 4; ++q) sPrev[r * 32 + c0 + q] = aembh[2 * 32 + c0 + q];
  }
  __syncthreads();

  v8f c0r[2][2], c1r[2][2];
  {
    v8f z = {};
#pragma unroll
    for (int mt = 0; mt < 2; ++mt)
#pragma unroll
      for (int j = 0; j < 2; ++j) { c0r[mt][j] = z; c1r[mt][j] = z; }
  }
  float tlp = 0.f, tent = 0.f;

  for (int d = 0; d < 32; ++d) {
    // ================= layer 0 gates =================
    v8f acc[2][4][2];
    {
      const float* gb = Gbase0 + (size_t)rowBase * 1024;
#pragma unroll
      for (int mt = 0; mt < 2; ++mt)
#pragma unroll
        for (int G = 0; G < 4; ++G)
#pragma unroll
          for (int j = 0; j < 2; ++j) {
            int nt = G * 16 + wv * 2 + j;
            int col = nt * 16 + ln15;
            int rb = mt * 16 + rsub;
            v8f a;
#pragma unroll
            for (int r = 0; r < 8; ++r)
              a[r] = gb[(size_t)(rb + r) * 1024 + col];
            acc[mt][G][j] = a;
          }
    }
    {  // K-chunk 0: prev action embedding vs Wih0[:,256:288]
      v16h a0 = load_frag(sPrev, 32, 0, 0, lane);
      v16h a1 = load_frag(sPrev, 32, 16, 0, lane);
#pragma unroll
      for (int G = 0; G < 4; ++G)
#pragma unroll
        for (int j = 0; j < 2; ++j) {
          int nt = G * 16 + wv * 2 + j;
          v16h b = load_frag(Wih0h, 288, nt * 16, 256, lane);
          acc[0][G][j] = wmma16(a0, b, acc[0][G][j]);
          acc[1][G][j] = wmma16(a1, b, acc[1][G][j]);
        }
    }
    for (int kc = 0; kc < 8; ++kc) {  // h0 @ Whh0^T
      v16h a0 = load_frag(sH0, 256, 0, kc * 32, lane);
      v16h a1 = load_frag(sH0, 256, 16, kc * 32, lane);
#pragma unroll
      for (int G = 0; G < 4; ++G)
#pragma unroll
        for (int j = 0; j < 2; ++j) {
          int nt = G * 16 + wv * 2 + j;
          v16h b = load_frag(Whh0h, 256, nt * 16, kc * 32, lane);
          acc[0][G][j] = wmma16(a0, b, acc[0][G][j]);
          acc[1][G][j] = wmma16(a1, b, acc[1][G][j]);
        }
    }
    __syncthreads();  // all waves done reading old sH0 / sPrev
#pragma unroll
    for (int mt = 0; mt < 2; ++mt)
#pragma unroll
      for (int j = 0; j < 2; ++j) {
        v8f iv = acc[mt][0][j], fv = acc[mt][1][j];
        v8f gv = acc[mt][2][j], ov = acc[mt][3][j];
        v8f c = c0r[mt][j];
#pragma unroll
        for (int r = 0; r < 8; ++r) {
          float cn = sigm(fv[r]) * c[r] + sigm(iv[r]) * tanhf(gv[r]);
          float hn = sigm(ov[r]) * tanhf(cn);
          c[r] = cn;
          sH0[(mt * 16 + rsub + r) * 256 + colb + j * 16 + ln15] = (_Float16)hn;
        }
        c0r[mt][j] = c;
      }
    __syncthreads();  // new h0 visible

    // ================= layer 1 gates =================
    v8f acc1[2][4][2];
#pragma unroll
    for (int G = 0; G < 4; ++G)
#pragma unroll
      for (int j = 0; j < 2; ++j) {
        int n = (G * 16 + wv * 2 + j) * 16 + ln15;
        float bv = b_ih1[n] + b_hh1[n];
        v8f a;
#pragma unroll
        for (int r = 0; r < 8; ++r) a[r] = bv;
        acc1[0][G][j] = a;
        acc1[1][G][j] = a;
      }
    for (int kc = 0; kc < 8; ++kc) {  // h0_new @ Wih1^T
      v16h a0 = load_frag(sH0, 256, 0, kc * 32, lane);
      v16h a1 = load_frag(sH0, 256, 16, kc * 32, lane);
#pragma unroll
      for (int G = 0; G < 4; ++G)
#pragma unroll
        for (int j = 0; j < 2; ++j) {
          int nt = G * 16 + wv * 2 + j;
          v16h b = load_frag(Wih1h, 256, nt * 16, kc * 32, lane);
          acc1[0][G][j] = wmma16(a0, b, acc1[0][G][j]);
          acc1[1][G][j] = wmma16(a1, b, acc1[1][G][j]);
        }
    }
    for (int kc = 0; kc < 8; ++kc) {  // h1_old @ Whh1^T
      v16h a0 = load_frag(sH1, 256, 0, kc * 32, lane);
      v16h a1 = load_frag(sH1, 256, 16, kc * 32, lane);
#pragma unroll
      for (int G = 0; G < 4; ++G)
#pragma unroll
        for (int j = 0; j < 2; ++j) {
          int nt = G * 16 + wv * 2 + j;
          v16h b = load_frag(Whh1h, 256, nt * 16, kc * 32, lane);
          acc1[0][G][j] = wmma16(a0, b, acc1[0][G][j]);
          acc1[1][G][j] = wmma16(a1, b, acc1[1][G][j]);
        }
    }
    __syncthreads();  // all waves done reading old sH1
#pragma unroll
    for (int mt = 0; mt < 2; ++mt)
#pragma unroll
      for (int j = 0; j < 2; ++j) {
        v8f iv = acc1[mt][0][j], fv = acc1[mt][1][j];
        v8f gv = acc1[mt][2][j], ov = acc1[mt][3][j];
        v8f c = c1r[mt][j];
#pragma unroll
        for (int r = 0; r < 8; ++r) {
          float cn = sigm(fv[r]) * c[r] + sigm(iv[r]) * tanhf(gv[r]);
          float hn = sigm(ov[r]) * tanhf(cn);
          c[r] = cn;
          sH1[(mt * 16 + rsub + r) * 256 + colb + j * 16 + ln15] = (_Float16)hn;
        }
        c1r[mt][j] = c;
      }
    __syncthreads();  // new h1 visible

    // ================= head: logits, log-softmax, entropy =================
    {
      int r = t >> 3, g = t & 7;
      const float* Wd = headsW + (size_t)d * 512;  // [256,2]
      float p0 = 0.f, p1 = 0.f;
      int k0 = g * 32;
      for (int k = 0; k < 32; ++k) {
        float hv = (float)sH1[r * 256 + k0 + k];
        p0 += hv * Wd[(k0 + k) * 2 + 0];
        p1 += hv * Wd[(k0 + k) * 2 + 1];
      }
      p0 += __shfl_xor(p0, 1); p0 += __shfl_xor(p0, 2); p0 += __shfl_xor(p0, 4);
      p1 += __shfl_xor(p1, 1); p1 += __shfl_xor(p1, 2); p1 += __shfl_xor(p1, 4);
      if (g == 0) {
        float l0 = p0 + headsB[d * 2 + 0];
        float l1 = p1 + headsB[d * 2 + 1];
        float mx = fmaxf(l0, l1);
        float lse = mx + __logf(__expf(l0 - mx) + __expf(l1 - mx));
        float lp0 = l0 - lse, lp1 = l1 - lse;
        float e0 = __expf(lp0), e1 = __expf(lp1);
        float ent = -(e0 * lp0 + e1 * lp1);
        int a = actions[(size_t)(rowBase + r) * 32 + d];
        size_t oi = (size_t)(rowBase + r) * 32 + d;
        prefix[oi] = tlp;   // prefix lp BEFORE adding this step
        perent[oi] = ent;
        tlp += (a == 0) ? lp0 : lp1;
        tent += ent;
        sAct[r] = a;
      }
    }
    __syncthreads();
    {  // prev <- aemb[action]
      int r = t >> 3, c0 = (t & 7) * 4;
      int a = sAct[r];
#pragma unroll
      for (int q = 0; q < 4; ++q) sPrev[r * 32 + c0 + q] = aembh[a * 32 + c0 + q];
    }
    __syncthreads();
  }

  if ((t & 7) == 0) {
    int r = t >> 3;
    tot_lp[rowBase + r] = tlp;
    tot_ent[rowBase + r] = tent;
  }
}

// ---------------------------------------------------------------------------
// launch
// ---------------------------------------------------------------------------
extern "C" void kernel_launch(void* const* d_in, const int* in_sizes, int n_in,
                              void* d_out, int out_size, void* d_ws,
                              size_t ws_size, hipStream_t stream) {
  const float* states = (const float*)d_in[0];
  const int*   actions = (const int*)d_in[1];
  const float* se_W  = (const float*)d_in[2];
  const float* se_b  = (const float*)d_in[3];
  const float* aemb  = (const float*)d_in[4];
  const float* W_ih0 = (const float*)d_in[5];
  const float* W_hh0 = (const float*)d_in[6];
  const float* b_ih0 = (const float*)d_in[7];
  const float* b_hh0 = (const float*)d_in[8];
  const float* W_ih1 = (const float*)d_in[9];
  const float* W_hh1 = (const float*)d_in[10];
  const float* b_ih1 = (const float*)d_in[11];
  const float* b_hh1 = (const float*)d_in[12];
  const float* headsW = (const float*)d_in[13];
  const float* headsB = (const float*)d_in[14];

  char* ws = (char*)d_ws;
  size_t off = 0;
  auto take = [&](size_t bytes) {
    char* p = ws + off;
    off += (bytes + 255) & ~(size_t)255;
    return p;
  };
  _Float16* states16 = (_Float16*)take((size_t)8192 * 256 * 2);
  _Float16* seW16    = (_Float16*)take((size_t)256 * 256 * 2);
  _Float16* semb16   = (_Float16*)take((size_t)8192 * 256 * 2);
  _Float16* Wih0h    = (_Float16*)take((size_t)1024 * 288 * 2);
  _Float16* Whh0h    = (_Float16*)take((size_t)1024 * 256 * 2);
  _Float16* Wih1h    = (_Float16*)take((size_t)1024 * 256 * 2);
  _Float16* Whh1h    = (_Float16*)take((size_t)1024 * 256 * 2);
  _Float16* aembh    = (_Float16*)take((size_t)3 * 32 * 2);
  float*    Gbase0   = (float*)take((size_t)8192 * 1024 * 4);

  auto cvt = [&](const float* src, _Float16* dst, int n) {
    cvt_f32_f16<<<(n + 255) / 256, 256, 0, stream>>>(src, dst, n);
  };
  cvt(states, states16, 8192 * 256);
  cvt(se_W, seW16, 256 * 256);
  cvt(W_ih0, Wih0h, 1024 * 288);
  cvt(W_hh0, Whh0h, 1024 * 256);
  cvt(W_ih1, Wih1h, 1024 * 256);
  cvt(W_hh1, Whh1h, 1024 * 256);
  cvt(aemb, aembh, 3 * 32);

  semb_gemm<<<1024, 256, 0, stream>>>(states16, seW16, se_b, semb16);
  gbase_gemm<<<4096, 256, 0, stream>>>(semb16, Wih0h, b_ih0, b_hh0, Gbase0);

  float* out = (float*)d_out;
  float* tot_lp  = out;
  float* tot_ent = out + 8192;
  float* prefix  = out + 16384;
  float* perent  = out + 16384 + 8192 * 32;

  actor_recurrent<<<256, 256, 0, stream>>>(Wih0h, Whh0h, Wih1h, Whh1h, Gbase0,
                                           b_ih1, b_hh1, aembh, actions, headsW,
                                           headsB, tot_lp, tot_ent, prefix,
                                           perent);
}
